// PGenGraphModel_23055384445125
// MI455X (gfx1250) — matrix operand order repaired and checked by reading, not compiled, not run on hardware
//
#include <hip/hip_runtime.h>
#include <hip/hip_bf16.h>
#include <cstdint>

// ---------------------------------------------------------------------------
// Problem constants (match reference)
// ---------------------------------------------------------------------------
#define N_NODES  100000
#define N_EDGES  400000
#define N_GRAPHS 2048
#define F_IN     128
#define HID      256
#define EMB      128
#define GENE_IN  1280

typedef __attribute__((ext_vector_type(16))) __bf16 v16bf;
typedef __attribute__((ext_vector_type(8)))  __bf16 v8bf;
typedef __attribute__((ext_vector_type(8)))  float  v8f;

// ---------------------------------------------------------------------------
// Helpers
// ---------------------------------------------------------------------------
__device__ __forceinline__ int f2ord(float f) {
  int i = __float_as_int(f);
  return (i >= 0) ? i : (i ^ 0x7fffffff);   // monotonic float->int map (involution)
}
__device__ __forceinline__ float ord2f(int i) {
  return __int_as_float((i >= 0) ? i : (i ^ 0x7fffffff));
}
__device__ __forceinline__ float gelu_exact(float x) {
  return 0.5f * x * (1.0f + erff(x * 0.7071067811865475f));
}

// ---------------------------------------------------------------------------
// WMMA GEMM: C[M x N] = A[M x K] * B[K x N] (+ bias), bf16 in / f32 acc.
// Block tile 64x64, 8 waves, each wave computes a 16x32 strip via two
// v_wmma_f32_16x16x32_bf16 per K-step.
// Requirements: K % 32 == 0, N % 64 == 0, A rows 16B-aligned (K % 4 == 0).
//
// LDS is stored FRAGMENT-READY: staging threads write each converted bf16
// into the exact slot its consuming lane reads, so fragment loads are
// contiguous 2 x ds_load_b128 per matrix per lane (no scalar u16 gathers).
//   A slot: As[wmTile][lane][i], lane = (kg&1)*16 + row%16, i = (kg>>1)*8 + j
//   B slot: Bs[wnTile][lane][i], lane = (kq&1)*16 + col%16, i = (kq>>1)*8 + j
// (kg/kq index groups of 8 consecutive k; j in 0..7) — per ISA 7.12.2 layout.
// ---------------------------------------------------------------------------
__global__ __launch_bounds__(256)
void pg_wmma_gemm(const float* __restrict__ A, const float* __restrict__ B,
                  const float* __restrict__ bias, float* __restrict__ C,
                  int M, int N, int K, int ldc)
{
  __shared__ __align__(16) __bf16 As[4][32][16];   // 4 wave-tile rows
  __shared__ __align__(16) __bf16 Bs[4][32][16];   // 4 wave-tile cols

  const int t    = threadIdx.x;
  const int lane = t & 31;
  const int wave = t >> 5;          // 0..7
  const int wm   = wave & 3;        // wave tile row (0..3)
  const int wn0  = (wave >> 2) * 2; // wave tile col pair base (0 or 2)

  const int blockM = blockIdx.y * 64;
  const int blockN = blockIdx.x * 64;

  // staging roles (fixed per thread)
  const int ar  = t >> 2;                 // A row 0..63
  const int akg = t & 3;                  // A k-group of 8
  const int aLane = (akg & 1) * 16 + (ar & 15);
  const int aOff  = (akg >> 1) * 8;
  const int aTile = ar >> 4;
  const int agr = blockM + ar;
  const bool aOk = (agr < M);

  const int bc  = t & 63;                 // B col 0..63
  const int bkq = t >> 6;                 // B k-group of 8
  const int bLane = (bkq & 1) * 16 + (bc & 15);
  const int bOff  = (bkq >> 1) * 8;
  const int bTile = bc >> 4;
  const int bgc = blockN + bc;

  v8f acc0 = {};
  v8f acc1 = {};

  for (int k0 = 0; k0 < K; k0 += 32) {
    // ---- stage A: 8 consecutive floats of one row -> one ds_store_b128 ----
    {
      v8bf ab = {};
      if (aOk) {
        const float* ap = A + (size_t)agr * K + k0 + akg * 8;
        float4 v0 = *(const float4*)ap;
        float4 v1 = *(const float4*)(ap + 4);
        ab[0] = (__bf16)v0.x; ab[1] = (__bf16)v0.y;
        ab[2] = (__bf16)v0.z; ab[3] = (__bf16)v0.w;
        ab[4] = (__bf16)v1.x; ab[5] = (__bf16)v1.y;
        ab[6] = (__bf16)v1.z; ab[7] = (__bf16)v1.w;
      }
      *(v8bf*)&As[aTile][aLane][aOff] = ab;
    }
    // ---- stage B: 8 k's of one column (coalesced across threads) ----
    {
      const float* bp = B + (size_t)(k0 + bkq * 8) * N + bgc;
      v8bf bb;
#pragma unroll
      for (int j = 0; j < 8; ++j) bb[j] = (__bf16)bp[(size_t)j * N];
      *(v8bf*)&Bs[bTile][bLane][bOff] = bb;
    }
    __syncthreads();

    // ---- fragment loads: contiguous 2 x ds_load_b128 per matrix ----
    v8bf a0 = *(const v8bf*)&As[wm][lane][0];
    v8bf a1 = *(const v8bf*)&As[wm][lane][8];
    v16bf afrag = __builtin_shufflevector(a0, a1, 0, 1, 2, 3, 4, 5, 6, 7,
                                          8, 9, 10, 11, 12, 13, 14, 15);
    v8bf b00 = *(const v8bf*)&Bs[wn0][lane][0];
    v8bf b01 = *(const v8bf*)&Bs[wn0][lane][8];
    v16bf bfrag0 = __builtin_shufflevector(b00, b01, 0, 1, 2, 3, 4, 5, 6, 7,
                                           8, 9, 10, 11, 12, 13, 14, 15);
    v8bf b10 = *(const v8bf*)&Bs[wn0 + 1][lane][0];
    v8bf b11 = *(const v8bf*)&Bs[wn0 + 1][lane][8];
    v16bf bfrag1 = __builtin_shufflevector(b10, b11, 0, 1, 2, 3, 4, 5, 6, 7,
                                           8, 9, 10, 11, 12, 13, 14, 15);

    acc0 = __builtin_amdgcn_wmma_f32_16x16x32_bf16(false, afrag, false, bfrag0,
                                                   (short)0, acc0, false, false);
    acc1 = __builtin_amdgcn_wmma_f32_16x16x32_bf16(false, afrag, false, bfrag1,
                                                   (short)0, acc1, false, false);
    __syncthreads();
  }

  // C/D layout: VGPR r -> row = (lane>>4)*8 + r, col = lane&15
  const int half = lane >> 4;
  const int l16  = lane & 15;
#pragma unroll
  for (int r = 0; r < 8; ++r) {
    int gm = blockM + wm * 16 + half * 8 + r;
    if (gm < M) {
      int gn0 = blockN + wn0 * 16 + l16;
      int gn1 = gn0 + 16;
      C[(size_t)gm * ldc + gn0] = acc0[r] + (bias ? bias[gn0] : 0.0f);
      C[(size_t)gm * ldc + gn1] = acc1[r] + (bias ? bias[gn1] : 0.0f);
    }
  }
}

// ---------------------------------------------------------------------------
// Elementwise / fill kernels
// ---------------------------------------------------------------------------
__global__ void pg_fill_f32(float* p, float v, int n) {
  int i = blockIdx.x * blockDim.x + threadIdx.x;
  if (i < n) p[i] = v;
}
__global__ void pg_fill_i32(int* p, int v, int n) {
  int i = blockIdx.x * blockDim.x + threadIdx.x;
  if (i < n) p[i] = v;
}
__global__ void pg_bias_bcast(float* out, const float* __restrict__ b, int rows, int cols) {
  int i = blockIdx.x * blockDim.x + threadIdx.x;
  if (i < rows * cols) out[i] = b[i % cols];
}
__global__ void pg_elu(float* p, int n) {
  int i = blockIdx.x * blockDim.x + threadIdx.x;
  if (i < n) { float x = p[i]; p[i] = (x > 0.0f) ? x : expm1f(x); }
}
__global__ void pg_gelu(float* p, int n) {
  int i = blockIdx.x * blockDim.x + threadIdx.x;
  if (i < n) p[i] = gelu_exact(p[i]);
}
__global__ void pg_bn_gelu(float* p, const float* __restrict__ g, const float* __restrict__ b,
                           const float* __restrict__ mean, const float* __restrict__ var, int n) {
  int i = blockIdx.x * blockDim.x + threadIdx.x;
  if (i < n) {
    int c = i & (HID - 1);
    float z = (p[i] - mean[c]) * rsqrtf(var[c] + 1e-5f) * g[c] + b[c];
    p[i] = gelu_exact(z);
  }
}

// ---------------------------------------------------------------------------
// Attention scores: a_src[n] = h[n,:].att_src ; a_dst[n] = h[n,:].att_dst
// ---------------------------------------------------------------------------
__global__ __launch_bounds__(256)
void pg_att_dot(const float* __restrict__ h, const float* __restrict__ att_s,
                const float* __restrict__ att_d, float* a_s, float* a_d) {
  int wid  = (blockIdx.x * blockDim.x + threadIdx.x) >> 5;
  int lane = threadIdx.x & 31;
  if (wid >= N_NODES) return;
  const float* row = h + (size_t)wid * HID;
  float ss = 0.0f, sd = 0.0f;
#pragma unroll
  for (int j = 0; j < HID / 32; ++j) {
    int c = lane + 32 * j;
    float x = row[c];
    ss += x * att_s[c];
    sd += x * att_d[c];
  }
  for (int o = 16; o > 0; o >>= 1) {
    ss += __shfl_down(ss, o, 32);
    sd += __shfl_down(sd, o, 32);
  }
  if (lane == 0) { a_s[wid] = ss; a_d[wid] = sd; }
}

// ---------------------------------------------------------------------------
// Edge passes (virtual edges [E, E+N) are self-loops).
// ---------------------------------------------------------------------------
__global__ void pg_edge_max(const int* __restrict__ ei, const float* __restrict__ a_s,
                            const float* __restrict__ a_d, float* eval, int* mint) {
  int i = blockIdx.x * blockDim.x + threadIdx.x;
  if (i >= N_EDGES + N_NODES) return;
  int s, d;
  if (i < N_EDGES) { s = ei[i]; d = ei[N_EDGES + i]; } else { s = d = i - N_EDGES; }
  float v = a_s[s] + a_d[d];
  v = (v > 0.0f) ? v : 0.2f * v;          // leaky_relu(0.2)
  eval[i] = v;
  atomicMax(&mint[d], f2ord(v));
}
__global__ void pg_edge_exp(const int* __restrict__ ei, float* eval,
                            const int* __restrict__ mint, float* denom) {
  int i = blockIdx.x * blockDim.x + threadIdx.x;
  if (i >= N_EDGES + N_NODES) return;
  int d = (i < N_EDGES) ? ei[N_EDGES + i] : (i - N_EDGES);
  float ex = expf(eval[i] - ord2f(mint[d]));
  eval[i] = ex;
  atomicAdd(&denom[d], ex);
}
__global__ __launch_bounds__(256)
void pg_edge_aggr(const int* __restrict__ ei, const float* __restrict__ eval,
                  const float* __restrict__ denom, const float* __restrict__ h,
                  float* out) {
  int wid  = (blockIdx.x * blockDim.x + threadIdx.x) >> 5;
  int lane = threadIdx.x & 31;
  if (wid >= N_EDGES + N_NODES) return;
  int s, d;
  if (wid < N_EDGES) { s = ei[wid]; d = ei[N_EDGES + wid]; } else { s = d = wid - N_EDGES; }
  float alpha = eval[wid] / denom[d];
  const float* hs = h + (size_t)s * HID;
  float* od = out + (size_t)d * HID;
#pragma unroll
  for (int j = 0; j < HID / 32; ++j) {
    int c = lane + 32 * j;
    atomicAdd(&od[c], alpha * hs[c]);     // out (102 MB) lives in the 192 MB L2
  }
}

// ---------------------------------------------------------------------------
// Graph pooling (uses the real batch array): sum + ordered-int max + counts.
// ---------------------------------------------------------------------------
__global__ __launch_bounds__(256)
void pg_pool_scatter(const float* __restrict__ h, const int* __restrict__ batch,
                     float* psum, int* pmax, int* cnt) {
  int n = blockIdx.x;       // one node per block
  int c = threadIdx.x;      // channel
  int g = batch[n];
  float v = h[(size_t)n * HID + c];
  atomicAdd(&psum[g * HID + c], v);
  atomicMax(&pmax[g * HID + c], f2ord(v));
  if (c == 0) atomicAdd(&cnt[g], 1);
}
__global__ void pg_pool_combine(float* psum /*in: sum, out: mean+max*/,
                                const int* __restrict__ pmax, const int* __restrict__ cnt) {
  int i = blockIdx.x * blockDim.x + threadIdx.x;
  if (i >= N_GRAPHS * HID) return;
  int g = i >> 8;
  psum[i] = psum[i] / (float)cnt[g] + ord2f(pmax[i]);
}

// ---------------------------------------------------------------------------
// LayerNorm: blockDim.x == dim (128 or 256), one row per block.
// ---------------------------------------------------------------------------
__global__ void pg_layernorm(const float* __restrict__ in, const float* __restrict__ g,
                             const float* __restrict__ b, float* out, int dim, int ostride) {
  __shared__ float sm[256];
  int r = blockIdx.x, c = threadIdx.x;
  float x = in[(size_t)r * dim + c];
  sm[c] = x; __syncthreads();
  for (int s = dim >> 1; s > 0; s >>= 1) { if (c < s) sm[c] += sm[c + s]; __syncthreads(); }
  float mu = sm[0] / (float)dim; __syncthreads();
  float dx = x - mu;
  sm[c] = dx * dx; __syncthreads();
  for (int s = dim >> 1; s > 0; s >>= 1) { if (c < s) sm[c] += sm[c + s]; __syncthreads(); }
  float var = sm[0] / (float)dim;
  out[(size_t)r * ostride + c] = dx * rsqrtf(var + 1e-5f) * g[c] + b[c];
}

// ---------------------------------------------------------------------------
// Heads: one wave per (graph, out-dim) element.
// ---------------------------------------------------------------------------
__global__ __launch_bounds__(256)
void pg_head(const float* __restrict__ fused, const float* __restrict__ W,
             const float* __restrict__ b, float* out, int D) {
  int wid  = (blockIdx.x * blockDim.x + threadIdx.x) >> 5;
  int lane = threadIdx.x & 31;
  if (wid >= N_GRAPHS * D) return;
  int g = wid / D, d = wid % D;
  const float* row = fused + (size_t)g * HID;
  float s = 0.0f;
#pragma unroll
  for (int j = 0; j < HID / 32; ++j) {
    int c = lane + 32 * j;
    s += row[c] * W[c * D + d];
  }
  for (int o = 16; o > 0; o >>= 1) s += __shfl_down(s, o, 32);
  if (lane == 0) out[g * D + d] = s + b[d];
}

// ---------------------------------------------------------------------------
// Host orchestration
// ---------------------------------------------------------------------------
static inline int cdiv(int a, int b) { return (a + b - 1) / b; }

extern "C" void kernel_launch(void* const* d_in, const int* in_sizes, int n_in,
                              void* d_out, int out_size, void* d_ws, size_t ws_size,
                              hipStream_t stream) {
  (void)in_sizes; (void)n_in; (void)out_size; (void)ws_size;

  // Inputs in setup_inputs() insertion order, params flattened recursively.
  const float* x        = (const float*)d_in[0];   // [100000,128]
  const int*   ei       = (const int*)  d_in[1];   // [2,400000]
  const int*   batch    = (const int*)  d_in[2];   // [100000]
  const float* gene_in  = (const float*)d_in[3];   // [2048,1280]
  const float* gatW[3]  = {(const float*)d_in[4],  (const float*)d_in[8],  (const float*)d_in[12]};
  const float* gatAS[3] = {(const float*)d_in[5],  (const float*)d_in[9],  (const float*)d_in[13]};
  const float* gatAD[3] = {(const float*)d_in[6],  (const float*)d_in[10], (const float*)d_in[14]};
  const float* gatB[3]  = {(const float*)d_in[7],  (const float*)d_in[11], (const float*)d_in[15]};
  const float* outW   = (const float*)d_in[16];    // [256,128]
  const float* outB   = (const float*)d_in[17];    // [128]
  const float* gW1    = (const float*)d_in[18];    // [1280,256]
  const float* gb1    = (const float*)d_in[19];
  const float* ln1g   = (const float*)d_in[20];
  const float* ln1b   = (const float*)d_in[21];
  const float* gW2    = (const float*)d_in[22];    // [256,128]
  const float* gb2    = (const float*)d_in[23];
  const float* ln2g   = (const float*)d_in[24];
  const float* ln2b   = (const float*)d_in[25];
  const float* fW     = (const float*)d_in[26];    // [256,256]
  const float* fb     = (const float*)d_in[27];
  const float* bng    = (const float*)d_in[28];
  const float* bnb    = (const float*)d_in[29];
  const float* bnmean = (const float*)d_in[30];
  const float* bnvar  = (const float*)d_in[31];
  const float* hW[3]  = {(const float*)d_in[32], (const float*)d_in[34], (const float*)d_in[36]};
  const float* hB[3]  = {(const float*)d_in[33], (const float*)d_in[35], (const float*)d_in[37]};
  const int    hD[3]  = {2, 5, 3};

  float* out = (float*)d_out;                       // 2048*2 | 2048*5 | 2048*3

  // workspace partition (floats); offsets keep 16B alignment for float4 loads
  float* ws    = (float*)d_ws;
  float* h_buf = ws;                                  // [100000,256]
  float* g_buf = h_buf + (size_t)N_NODES * HID;       // [100000,256]
  float* a_s   = g_buf + (size_t)N_NODES * HID;       // [100000]
  float* a_d   = a_s + N_NODES;                       // [100000]
  int*   mint  = (int*)(a_d + N_NODES);               // [100000]
  float* denom = (float*)(mint + N_NODES);            // [100000]
  float* ebuf  = denom + N_NODES;                     // [500000]; later gene tmp
  float* psum  = ebuf + (N_EDGES + N_NODES);          // [2048,256] -> pooled
  int*   pmax  = (int*)(psum + (size_t)N_GRAPHS * HID);
  int*   cnt   = pmax + (size_t)N_GRAPHS * HID;       // [2048]
  float* comb  = (float*)(cnt + N_GRAPHS);            // [2048,256] concat(drug, gene)
  float* gh    = comb + (size_t)N_GRAPHS * HID;       // [2048,256] gene hidden -> fused

  const int ET = N_EDGES + N_NODES;                   // edges + self-loops
  const int NH = N_NODES * HID;

  // ----------------- GAT layers -----------------
  const float* xin = x;
  int Kl = F_IN;
  for (int l = 0; l < 3; ++l) {
    pg_wmma_gemm<<<dim3(HID / 64, cdiv(N_NODES, 64)), 256, 0, stream>>>(
        xin, gatW[l], nullptr, g_buf, N_NODES, HID, Kl, HID);
    pg_att_dot<<<cdiv(N_NODES * 32, 256), 256, 0, stream>>>(g_buf, gatAS[l], gatAD[l], a_s, a_d);
    pg_fill_i32<<<cdiv(N_NODES, 256), 256, 0, stream>>>(mint, INT_MIN, N_NODES);
    pg_fill_f32<<<cdiv(N_NODES, 256), 256, 0, stream>>>(denom, 0.0f, N_NODES);
    pg_edge_max<<<cdiv(ET, 256), 256, 0, stream>>>(ei, a_s, a_d, ebuf, mint);
    pg_edge_exp<<<cdiv(ET, 256), 256, 0, stream>>>(ei, ebuf, mint, denom);
    pg_bias_bcast<<<cdiv(NH, 256), 256, 0, stream>>>(h_buf, gatB[l], N_NODES, HID);
    pg_edge_aggr<<<cdiv(ET * 32, 256), 256, 0, stream>>>(ei, ebuf, denom, g_buf, h_buf);
    pg_elu<<<cdiv(NH, 256), 256, 0, stream>>>(h_buf, NH);
    xin = h_buf;
    Kl = HID;
  }

  // ----------------- pooling -----------------
  pg_fill_f32<<<cdiv(N_GRAPHS * HID, 256), 256, 0, stream>>>(psum, 0.0f, N_GRAPHS * HID);
  pg_fill_i32<<<cdiv(N_GRAPHS * HID, 256), 256, 0, stream>>>(pmax, INT_MIN, N_GRAPHS * HID);
  pg_fill_i32<<<cdiv(N_GRAPHS, 256), 256, 0, stream>>>(cnt, 0, N_GRAPHS);
  pg_pool_scatter<<<N_NODES, HID, 0, stream>>>(h_buf, batch, psum, pmax, cnt);
  pg_pool_combine<<<cdiv(N_GRAPHS * HID, 256), 256, 0, stream>>>(psum, pmax, cnt);

  // drug = pooled @ outW + outB  -> comb[:, 0:128]  (ldc = 256)
  pg_wmma_gemm<<<dim3(EMB / 64, cdiv(N_GRAPHS, 64)), 256, 0, stream>>>(
      psum, outW, outB, comb, N_GRAPHS, EMB, HID, HID);

  // ----------------- gene branch -----------------
  pg_wmma_gemm<<<dim3(HID / 64, cdiv(N_GRAPHS, 64)), 256, 0, stream>>>(
      gene_in, gW1, gb1, gh, N_GRAPHS, HID, GENE_IN, HID);
  pg_layernorm<<<N_GRAPHS, HID, 0, stream>>>(gh, ln1g, ln1b, gh, HID, HID);
  pg_gelu<<<cdiv(N_GRAPHS * HID, 256), 256, 0, stream>>>(gh, N_GRAPHS * HID);
  float* gtmp = ebuf;  // reuse edge buffer: 2048*128 <= 500000
  pg_wmma_gemm<<<dim3(EMB / 64, cdiv(N_GRAPHS, 64)), 256, 0, stream>>>(
      gh, gW2, gb2, gtmp, N_GRAPHS, EMB, HID, EMB);
  pg_layernorm<<<N_GRAPHS, EMB, 0, stream>>>(gtmp, ln2g, ln2b, comb + EMB, EMB, HID);

  // ----------------- fusion + heads -----------------
  pg_wmma_gemm<<<dim3(HID / 64, cdiv(N_GRAPHS, 64)), 256, 0, stream>>>(
      comb, fW, fb, gh, N_GRAPHS, HID, 2 * EMB, HID);       // fused (reuse gh)
  pg_bn_gelu<<<cdiv(N_GRAPHS * HID, 256), 256, 0, stream>>>(gh, bng, bnb, bnmean, bnvar,
                                                            N_GRAPHS * HID);

  float* outp = out;
  for (int i = 0; i < 3; ++i) {
    pg_head<<<cdiv(N_GRAPHS * hD[i] * 32, 256), 256, 0, stream>>>(gh, hW[i], hB[i], outp, hD[i]);
    outp += (size_t)N_GRAPHS * hD[i];
  }
}
